// LightweightMultiHeadSelfAttention_69252052681026
// MI455X (gfx1250) — matrix operand order repaired and checked by reading, not compiled, not run on hardware
//
#include <hip/hip_runtime.h>
#include <hip/hip_bf16.h>
#include <math.h>

typedef __attribute__((ext_vector_type(16))) __bf16 v16bf;
typedef __attribute__((ext_vector_type(8)))  __bf16 v8bf;
typedef __attribute__((ext_vector_type(8)))  float  v8f;
typedef __attribute__((ext_vector_type(8)))  unsigned short v8u;
typedef unsigned short u16;

#define HW_   12544
#define NP_   200704ll   // 16*112*112

__device__ __forceinline__ u16 f2bf(float f) {
  unsigned int u = __float_as_uint(f);
  u += 0x7FFFu + ((u >> 16) & 1u);          // round-to-nearest-even
  return (u16)(u >> 16);
}

__device__ __forceinline__ v8f zero_v8f() {
  v8f z;
#pragma unroll
  for (int i = 0; i < 8; ++i) z[i] = 0.f;
  return z;
}

__device__ __forceinline__ v8bf zero_v8bf() {
  v8bf z;
#pragma unroll
  for (int i = 0; i < 8; ++i) z[i] = (__bf16)0.f;
  return z;
}

__device__ __forceinline__ v16bf cat16(v8bf lo, v8bf hi) {
  return __builtin_shufflevector(lo, hi, 0,1,2,3,4,5,6,7,8,9,10,11,12,13,14,15);
}

__device__ __forceinline__ v8f wmma_bf16(v16bf a, v16bf b, v8f c) {
  return __builtin_amdgcn_wmma_f32_16x16x32_bf16(false, a, false, b, (short)0, c, false, false);
}

// A fragment (16x32 bf16) from row-major source; arow = pointer to this lane's row.
// Layout: lanes 0-15 hold K = k0+{0..15}; lanes 16-31 K = k0+{16..31}.
__device__ __forceinline__ v16bf load_a_frag(const u16* arow, int k0, int lane) {
  const int kb = (lane & 16) ? 16 : 0;
  v8bf lo = *reinterpret_cast<const v8bf*>(arow + k0 + kb);
  v8bf hi = *reinterpret_cast<const v8bf*>(arow + k0 + kb + 8);
  return cat16(lo, hi);
}

// ---------------------------------------------------------------------------
// prep: fold LN gamma into weight rows, reduce W*beta and W*gamma per row
// ---------------------------------------------------------------------------
__global__ void prep_rowA(const float* __restrict__ W, const float* __restrict__ g,
                          const float* __restrict__ bvec, u16* __restrict__ A,
                          float* __restrict__ wb, float* __restrict__ wg) {
  const int row = blockIdx.x, c = threadIdx.x;
  const float w = W[row * 256 + c];
  A[row * 256 + c] = f2bf(w * g[c]);
  __shared__ float sb[256], sg[256];
  sb[c] = w * bvec[c];
  sg[c] = w * g[c];
  __syncthreads();
  for (int s = 128; s > 0; s >>= 1) {
    if (c < s) { sb[c] += sb[c + s]; sg[c] += sg[c + s]; }
    __syncthreads();
  }
  if (c == 0) { wb[row] = sb[0]; wg[row] = sg[0]; }
}

__global__ void cvt_bf16(const float* __restrict__ src, u16* __restrict__ dst, int n) {
  int i = blockIdx.x * 256 + threadIdx.x;
  if (i < n) dst[i] = f2bf(src[i]);
}

// ---------------------------------------------------------------------------
// fused: per-pixel LayerNorm stats (mean, rstd) over 256 channels AND
// transpose [B,256,H,W] f32 -> [p][256] bf16, single pass over the tensor.
// grid NP/256, block 256 (one thread per pixel; lanes = consecutive pixels)
// ---------------------------------------------------------------------------
__global__ void fused_ln_transpose(const float* __restrict__ src, u16* __restrict__ dst,
                                   float* __restrict__ mean, float* __restrict__ rstd) {
  const long long p = (long long)blockIdx.x * 256 + threadIdx.x;
  const int b = (int)(p / HW_);
  const long long pp = p - (long long)b * HW_;
  const float* s = src + (size_t)b * 256 * HW_ + pp;
  u16* d = dst + (size_t)p * 256;
  float sum = 0.f, sq = 0.f;
#pragma unroll
  for (int i = 0; i < 32; ++i) {
    v8u t;
#pragma unroll
    for (int j = 0; j < 8; ++j) {
      float v = s[(size_t)(i * 8 + j) * HW_];
      sum += v; sq += v * v;
      t[j] = f2bf(v);
    }
    *reinterpret_cast<v8u*>(d + i * 8) = t;
  }
  const float mu = sum * (1.f / 256.f);
  const float var = sq * (1.f / 256.f) - mu * mu;
  mean[p] = mu;
  rstd[p] = rsqrtf(var + 1e-5f);
}

// ---------------------------------------------------------------------------
// QKV GEMM: qkvT[o][p] = bf16(rstd*acc + wb[o] - mean*rstd*wg[o]);  M=192,K=256
// grid (NP/64, 3), block 128 (4 waves); wave -> one 16(M) x 64(N) tile
// ---------------------------------------------------------------------------
__global__ void gemm_qkv(const u16* __restrict__ xT, const u16* __restrict__ A,
                         const float* __restrict__ wb, const float* __restrict__ wg,
                         const float* __restrict__ mean, const float* __restrict__ rstd,
                         u16* __restrict__ qkvT) {
  const int lane = threadIdx.x & 31;
  const int wave = threadIdx.x >> 5;
  const int m0 = (blockIdx.y * 4 + wave) * 16;
  const long long p0 = (long long)blockIdx.x * 64;
  const u16* arow = A + (size_t)(m0 + (lane & 15)) * 256;
  const int bsel = (lane & 16) ? 16 : 0;

  v8f acc[4];
#pragma unroll
  for (int n = 0; n < 4; ++n) acc[n] = zero_v8f();

  for (int k0 = 0; k0 < 256; k0 += 32) {
    v16bf a = load_a_frag(arow, k0, lane);
#pragma unroll
    for (int n = 0; n < 4; ++n) {
      const long long col = p0 + n * 16 + (lane & 15);
      const u16* bp = xT + (size_t)col * 256 + k0 + bsel;
      if (k0 + 32 < 256) __builtin_prefetch(bp + 32);
      v16bf b = *reinterpret_cast<const v16bf*>(bp);
      acc[n] = wmma_bf16(a, b, acc[n]);
    }
  }

  const int half = (lane & 16) ? 8 : 0;
  float wbv[8], wgv[8];
#pragma unroll
  for (int r = 0; r < 8; ++r) { wbv[r] = wb[m0 + r + half]; wgv[r] = wg[m0 + r + half]; }
#pragma unroll
  for (int n = 0; n < 4; ++n) {
    const long long col = p0 + n * 16 + (lane & 15);
    const float mu = mean[col], rs = rstd[col];
#pragma unroll
    for (int r = 0; r < 8; ++r) {
      float v = rs * acc[n][r] + wbv[r] - mu * rs * wgv[r];
      qkvT[(size_t)(m0 + r + half) * NP_ + col] = f2bf(v);
    }
  }
}

// ---------------------------------------------------------------------------
// windowed attention: one block (4 waves) per (window, head)
// qkvT rows: q = h*16+d, k = 64+h*16+d, v = 128+h*16+d
// ---------------------------------------------------------------------------
__global__ void attn_kernel(const u16* __restrict__ qkvT, u16* __restrict__ owT) {
  __shared__ __attribute__((aligned(16))) u16 qs[64 * 16];
  __shared__ __attribute__((aligned(16))) u16 ks[64 * 16];
  __shared__ __attribute__((aligned(16))) u16 vsT[16 * 64];
  __shared__ __attribute__((aligned(16))) u16 ps[64 * 72];

  const int u = blockIdx.x;
  const int h = u & 3;
  const int win = u >> 2;
  const int wx = win & 15, wy = (win >> 4) & 15, b = win >> 8;
  const long long pwin = (long long)b * HW_ + (long long)(wy * 7) * 112 + wx * 7;

  const int tid = threadIdx.x;
  const int lane = tid & 31;
  const int wv = tid >> 5;

  // stage q,k (row-major token x dim) and v transposed (dim x token), zero pad
  for (int i = tid; i < 1024; i += 128) {
    const int d = i >> 6, t = i & 63;
    u16 qv = 0, kv = 0, vv = 0;
    if (t < 49) {
      const long long p = pwin + (t / 7) * 112 + (t % 7);
      qv = qkvT[(size_t)(h * 16 + d) * NP_ + p];
      kv = qkvT[(size_t)(64 + h * 16 + d) * NP_ + p];
      vv = qkvT[(size_t)(128 + h * 16 + d) * NP_ + p];
    }
    qs[t * 16 + d] = qv;
    ks[t * 16 + d] = kv;
    vsT[d * 64 + t] = vv;
  }
  __syncthreads();

  const int row16 = lane & 15;
  const int half = (lane & 16) ? 8 : 0;
  const int qrow = wv * 16 + row16;

  // A fragment of q: K=dim (16 real, 16 zero-pad)
  v16bf aq;
  {
    v8bf lo = (lane < 16) ? *reinterpret_cast<const v8bf*>(&qs[qrow * 16]) : zero_v8bf();
    v8bf hi = (lane < 16) ? *reinterpret_cast<const v8bf*>(&qs[qrow * 16 + 8]) : zero_v8bf();
    aq = cat16(lo, hi);
  }

  // S = q * k^T  (4 N-tiles of 16 cols)
  v8f s[4];
#pragma unroll
  for (int n = 0; n < 4; ++n) s[n] = zero_v8f();
#pragma unroll
  for (int n = 0; n < 4; ++n) {
    const int j = n * 16 + row16;
    v16bf bk;
    if (lane < 16) bk = *reinterpret_cast<const v16bf*>(&ks[j * 16]);
    else           bk = cat16(zero_v8bf(), zero_v8bf());
    s[n] = wmma_bf16(aq, bk, s[n]);
  }

  // masked softmax per row (row's 16-per-tile values live across one lane half)
#pragma unroll
  for (int r = 0; r < 8; ++r) {
    float v[4];
#pragma unroll
    for (int n = 0; n < 4; ++n) {
      const int j = n * 16 + row16;
      v[n] = (j < 49) ? s[n][r] * 0.25f : -1e30f;
    }
    float m = fmaxf(fmaxf(v[0], v[1]), fmaxf(v[2], v[3]));
    m = fmaxf(m, __shfl_xor(m, 1, 32));
    m = fmaxf(m, __shfl_xor(m, 2, 32));
    m = fmaxf(m, __shfl_xor(m, 4, 32));
    m = fmaxf(m, __shfl_xor(m, 8, 32));
    float e[4], sum = 0.f;
#pragma unroll
    for (int n = 0; n < 4; ++n) {
      const int j = n * 16 + row16;
      e[n] = (j < 49) ? __expf(v[n] - m) : 0.f;
      sum += e[n];
    }
    sum += __shfl_xor(sum, 1, 32);
    sum += __shfl_xor(sum, 2, 32);
    sum += __shfl_xor(sum, 4, 32);
    sum += __shfl_xor(sum, 8, 32);
    const float inv = 1.f / sum;
    const int prow = wv * 16 + r + half;
#pragma unroll
    for (int n = 0; n < 4; ++n)
      ps[prow * 72 + n * 16 + row16] = f2bf(e[n] * inv);
  }

  // O = P * v   (K = 64 tokens, 2 WMMA steps); same-wave LDS RAW is in-order
  v8f oc = zero_v8f();
#pragma unroll
  for (int s2 = 0; s2 < 2; ++s2) {
    const int kb = (lane & 16) ? 16 : 0;
    const u16* pr = &ps[(wv * 16 + row16) * 72 + s2 * 32 + kb];
    v8bf lo = *reinterpret_cast<const v8bf*>(pr);
    v8bf hi = *reinterpret_cast<const v8bf*>(pr + 8);
    v16bf ap = cat16(lo, hi);
    v16bf bv = *reinterpret_cast<const v16bf*>(&vsT[row16 * 64 + s2 * 32 + ((lane & 16) ? 16 : 0)]);
    oc = wmma_bf16(ap, bv, oc);
  }

  // store O (cols = head dim, rows = tokens) -> owT[p][h*16+d]
#pragma unroll
  for (int r = 0; r < 8; ++r) {
    const int t = wv * 16 + r + half;
    if (t < 49) {
      const long long p = pwin + (t / 7) * 112 + (t % 7);
      owT[(size_t)p * 64 + h * 16 + row16] = f2bf(oc[r]);
    }
  }
}

// ---------------------------------------------------------------------------
// proj GEMM + residual: d_out[c,p] = sum_r projA[c,r]*owT[p,r] + x[c,p]
// M=256, K=64; grid (NP/64, 4), block 128
// ---------------------------------------------------------------------------
__global__ void gemm_proj(const u16* __restrict__ owT, const u16* __restrict__ A,
                          const float* __restrict__ x, float* __restrict__ out) {
  const int lane = threadIdx.x & 31;
  const int wave = threadIdx.x >> 5;
  const int m0 = (blockIdx.y * 4 + wave) * 16;
  const long long p0 = (long long)blockIdx.x * 64;
  const u16* arow = A + (size_t)(m0 + (lane & 15)) * 64;
  const int bsel = (lane & 16) ? 16 : 0;

  v8f acc[4];
#pragma unroll
  for (int n = 0; n < 4; ++n) acc[n] = zero_v8f();
#pragma unroll
  for (int k0 = 0; k0 < 64; k0 += 32) {
    v16bf a = load_a_frag(arow, k0, lane);
#pragma unroll
    for (int n = 0; n < 4; ++n) {
      const long long col = p0 + n * 16 + (lane & 15);
      v16bf b = *reinterpret_cast<const v16bf*>(owT + (size_t)col * 64 + k0 + bsel);
      acc[n] = wmma_bf16(a, b, acc[n]);
    }
  }

  const int half = (lane & 16) ? 8 : 0;
  const int bidx = (int)(p0 / HW_);
#pragma unroll
  for (int n = 0; n < 4; ++n) {
    const long long col = p0 + n * 16 + (lane & 15);
    const long long pp = col - (long long)bidx * HW_;
#pragma unroll
    for (int r = 0; r < 8; ++r) {
      const int c = m0 + r + half;
      const size_t idx = ((size_t)bidx * 256 + c) * HW_ + pp;
      out[idx] = acc[n][r] + x[idx];
    }
  }
}

// ---------------------------------------------------------------------------
// FFN1 GEMM: h1T[p][r] = gelu(rstd2*acc + wb[r] - mean2*rstd2*wg[r] + b1[r])
// M=64 (4 waves), K=256; grid (NP/64, 1), block 128
// ---------------------------------------------------------------------------
__global__ void gemm_ffn1(const u16* __restrict__ outT, const u16* __restrict__ A,
                          const float* __restrict__ wb, const float* __restrict__ wg,
                          const float* __restrict__ mean, const float* __restrict__ rstd,
                          const float* __restrict__ b1, u16* __restrict__ h1T) {
  const int lane = threadIdx.x & 31;
  const int wave = threadIdx.x >> 5;
  const int m0 = wave * 16;
  const long long p0 = (long long)blockIdx.x * 64;
  const u16* arow = A + (size_t)(m0 + (lane & 15)) * 256;
  const int bsel = (lane & 16) ? 16 : 0;

  v8f acc[4];
#pragma unroll
  for (int n = 0; n < 4; ++n) acc[n] = zero_v8f();
  for (int k0 = 0; k0 < 256; k0 += 32) {
    v16bf a = load_a_frag(arow, k0, lane);
#pragma unroll
    for (int n = 0; n < 4; ++n) {
      const long long col = p0 + n * 16 + (lane & 15);
      const u16* bp = outT + (size_t)col * 256 + k0 + bsel;
      if (k0 + 32 < 256) __builtin_prefetch(bp + 32);
      v16bf b = *reinterpret_cast<const v16bf*>(bp);
      acc[n] = wmma_bf16(a, b, acc[n]);
    }
  }

  const int half = (lane & 16) ? 8 : 0;
  float wbv[8], wgv[8], b1v[8];
#pragma unroll
  for (int r = 0; r < 8; ++r) {
    wbv[r] = wb[m0 + r + half];
    wgv[r] = wg[m0 + r + half];
    b1v[r] = b1[m0 + r + half];
  }
#pragma unroll
  for (int n = 0; n < 4; ++n) {
    const long long col = p0 + n * 16 + (lane & 15);
    const float mu = mean[col], rs = rstd[col];
#pragma unroll
    for (int r = 0; r < 8; ++r) {
      float v = rs * acc[n][r] + wbv[r] - mu * rs * wgv[r] + b1v[r];
      float g = 0.5f * v * (1.f + erff(v * 0.70710678118654752f));   // exact GELU
      h1T[(size_t)col * 64 + m0 + r + half] = f2bf(g);
    }
  }
}

// ---------------------------------------------------------------------------
// FFN2 GEMM + bias + residual RMW: d_out[c,p] += acc + b2[c]
// M=256, K=64; grid (NP/64, 4), block 128
// ---------------------------------------------------------------------------
__global__ void gemm_ffn2(const u16* __restrict__ h1T, const u16* __restrict__ A,
                          const float* __restrict__ b2, float* __restrict__ out) {
  const int lane = threadIdx.x & 31;
  const int wave = threadIdx.x >> 5;
  const int m0 = (blockIdx.y * 4 + wave) * 16;
  const long long p0 = (long long)blockIdx.x * 64;
  const u16* arow = A + (size_t)(m0 + (lane & 15)) * 64;
  const int bsel = (lane & 16) ? 16 : 0;

  v8f acc[4];
#pragma unroll
  for (int n = 0; n < 4; ++n) acc[n] = zero_v8f();
#pragma unroll
  for (int k0 = 0; k0 < 64; k0 += 32) {
    v16bf a = load_a_frag(arow, k0, lane);
#pragma unroll
    for (int n = 0; n < 4; ++n) {
      const long long col = p0 + n * 16 + (lane & 15);
      v16bf b = *reinterpret_cast<const v16bf*>(h1T + (size_t)col * 64 + k0 + bsel);
      acc[n] = wmma_bf16(a, b, acc[n]);
    }
  }

  const int half = (lane & 16) ? 8 : 0;
  const int bidx = (int)(p0 / HW_);
  float b2v[8];
#pragma unroll
  for (int r = 0; r < 8; ++r) b2v[r] = b2[m0 + r + half];
#pragma unroll
  for (int n = 0; n < 4; ++n) {
    const long long col = p0 + n * 16 + (lane & 15);
    const long long pp = col - (long long)bidx * HW_;
#pragma unroll
    for (int r = 0; r < 8; ++r) {
      const int c = m0 + r + half;
      const size_t idx = ((size_t)bidx * 256 + c) * HW_ + pp;
      out[idx] = out[idx] + acc[n][r] + b2v[r];
    }
  }
}

// ---------------------------------------------------------------------------
extern "C" void kernel_launch(void* const* d_in, const int* in_sizes, int n_in,
                              void* d_out, int out_size, void* d_ws, size_t ws_size,
                              hipStream_t stream) {
  (void)in_sizes; (void)n_in; (void)out_size; (void)ws_size;
  const float* x      = (const float*)d_in[0];
  const float* qkv_w  = (const float*)d_in[1];
  const float* proj_w = (const float*)d_in[2];
  const float* ln1_g  = (const float*)d_in[3];
  const float* ln1_b  = (const float*)d_in[4];
  const float* ln2_g  = (const float*)d_in[5];
  const float* ln2_b  = (const float*)d_in[6];
  const float* ffn_w1 = (const float*)d_in[7];
  const float* ffn_b1 = (const float*)d_in[8];
  const float* ffn_w2 = (const float*)d_in[9];
  const float* ffn_b2 = (const float*)d_in[10];
  float* out = (float*)d_out;

  char* ws = (char*)d_ws;
  // bufA: 200704*256*2 = 102.76 MB  (xT, then owT, then outT — lifetimes disjoint)
  u16* bufA = (u16*)ws;
  u16* owT  = bufA;                      // [p][64] bf16, overlays dead xT
  // bufB: 192*200704*2 = 77.07 MB   (qkvT, later h1T)
  u16* qkvT = (u16*)(ws + (size_t)NP_ * 256 * 2);
  u16* h1T  = qkvT;
  float* stats = (float*)(ws + (size_t)NP_ * 256 * 2 + (size_t)192 * NP_ * 2);
  float* mean1 = stats;
  float* rstd1 = stats + NP_;
  float* mean2 = stats + 2 * NP_;
  float* rstd2 = stats + 3 * NP_;
  char* wp = (char*)(stats + 4 * NP_);
  u16*   qkvA  = (u16*)wp;   wp += 192 * 256 * 2;
  float* qwb   = (float*)wp; wp += 192 * 4;
  float* qwg   = (float*)wp; wp += 192 * 4;
  u16*   ffn1A = (u16*)wp;   wp += 64 * 256 * 2;
  float* fwb   = (float*)wp; wp += 64 * 4;
  float* fwg   = (float*)wp; wp += 64 * 4;
  u16*   projA = (u16*)wp;   wp += 256 * 64 * 2;
  u16*   ffn2A = (u16*)wp;   wp += 256 * 64 * 2;

  const int npx64  = (int)(NP_ / 64);    // 3136
  const int npx256 = (int)(NP_ / 256);   // 784

  // 1. weight prep
  prep_rowA<<<192, 256, 0, stream>>>(qkv_w, ln1_g, ln1_b, qkvA, qwb, qwg);
  prep_rowA<<<64, 256, 0, stream>>>(ffn_w1, ln2_g, ln2_b, ffn1A, fwb, fwg);
  cvt_bf16<<<(256 * 64 + 255) / 256, 256, 0, stream>>>(proj_w, projA, 256 * 64);
  cvt_bf16<<<(256 * 64 + 255) / 256, 256, 0, stream>>>(ffn_w2, ffn2A, 256 * 64);

  // 2. fused LN1 stats + x transpose to bf16 [p][c] (single pass over x)
  fused_ln_transpose<<<npx256, 256, 0, stream>>>(x, bufA, mean1, rstd1);

  // 3. QKV projection (fused LN1)
  gemm_qkv<<<dim3(npx64, 3), 128, 0, stream>>>(bufA, qkvA, qwb, qwg, mean1, rstd1, qkvT);

  // 4. windowed multi-head attention (16 windows^2 * 16 batch * 4 heads)
  attn_kernel<<<16384, 128, 0, stream>>>(qkvT, owT);

  // 5. output projection + residual -> d_out (f32, NCHW == [c,p])
  gemm_proj<<<dim3(npx64, 4), 128, 0, stream>>>(owT, projA, x, out);

  // 6. fused LN2 stats + transpose d_out to bf16 [p][c] (reuses bufA)
  fused_ln_transpose<<<npx256, 256, 0, stream>>>(out, bufA, mean2, rstd2);

  // 7. FFN
  gemm_ffn1<<<dim3(npx64, 1), 128, 0, stream>>>(bufA, ffn1A, fwb, fwg, mean2, rstd2, ffn_b1, h1T);
  gemm_ffn2<<<dim3(npx64, 4), 128, 0, stream>>>(h1T, ffn2A, ffn_b2, out);
}
